// MultiHeadSelfAttention_6682969113140
// MI455X (gfx1250) — matrix operand order repaired and checked by reading, not compiled, not run on hardware
//
#include <hip/hip_runtime.h>

#define S_LEN 4096
#define EMB   768
#define NH    12
#define DH    64

typedef __attribute__((ext_vector_type(16))) __bf16 v16bf;
typedef __attribute__((ext_vector_type(8)))  __bf16 v8bf;
typedef __attribute__((ext_vector_type(8)))  float  v8f;

static __device__ inline v8f wmma_bf16(v16bf a, v16bf b, v8f c) {
  // D(f32 16x16) = A(bf16 16x32) * B(bf16 32x16) + C
  return __builtin_amdgcn_wmma_f32_16x16x32_bf16(false, a, false, b, (short)0, c,
                                                 false, false);
}

// A-operand fragment (16x32, M = lane&15): lane holds K = {8g..8g+7, 16+8g..23+8g}
static __device__ inline v16bf a_frag_f32(const float* rowk, int g) {
  const float* p0 = rowk + 8 * g;
  const float* p1 = rowk + 16 + 8 * g;
  v16bf a;
#pragma unroll
  for (int i = 0; i < 8; ++i) { a[i] = (__bf16)p0[i]; a[8 + i] = (__bf16)p1[i]; }
  return a;
}

static __device__ inline v16bf a_frag_bf16(const __bf16* rowk, int g) {
  v8bf c0 = *(const v8bf*)(rowk + 8 * g);
  v8bf c1 = *(const v8bf*)(rowk + 16 + 8 * g);
  v16bf a;
#pragma unroll
  for (int i = 0; i < 8; ++i) { a[i] = c0[i]; a[8 + i] = c1[i]; }
  return a;
}

// B-operand fragment (32x16, N = lane&15): lane holds K = {8g..8g+7, 16+8g..23+8g}.
// colk points at Bt[n][k0] where Bt is the transposed (N-major) B matrix.
static __device__ inline v16bf b_frag_bf16(const __bf16* colk, int g) {
  v8bf c0 = *(const v8bf*)(colk + 8 * g);
  v8bf c1 = *(const v8bf*)(colk + 16 + 8 * g);
  v16bf b;
#pragma unroll
  for (int i = 0; i < 8; ++i) { b[i] = c0[i]; b[8 + i] = c1[i]; }
  return b;
}

// ---------------- Kernel 1: weight transpose + bf16 downconvert -------------
// Wt[z][n][k] = (bf16) W_z[k][n],  z in {q,k,v,o}; tiled for coalescing.
__global__ __launch_bounds__(256) void transpose_w_kernel(
    const float* __restrict__ Wq, const float* __restrict__ Wk,
    const float* __restrict__ Wv, const float* __restrict__ Wo,
    __bf16* __restrict__ Wt) {
  __shared__ float tile[32][33];
  const int z = blockIdx.z;
  const float* W = (z == 0) ? Wq : (z == 1) ? Wk : (z == 2) ? Wv : Wo;
  const int kb = blockIdx.x * 32;
  const int nb = blockIdx.y * 32;
  const int tx = threadIdx.x & 31;
  const int ty = threadIdx.x >> 5;   // 8 rows per pass
#pragma unroll
  for (int i = 0; i < 4; ++i)
    tile[ty + i * 8][tx] = W[(size_t)(kb + ty + i * 8) * EMB + nb + tx];
  __syncthreads();
  __bf16* dst = Wt + (size_t)z * EMB * EMB;
#pragma unroll
  for (int i = 0; i < 4; ++i)
    dst[(size_t)(nb + ty + i * 8) * EMB + kb + tx] = (__bf16)tile[tx][ty + i * 8];
}

// ---------------- Kernel 2: QKV projections ---------------------------------
// z=0: Q = queries@Wq -> Qh[h][S][64] bf16
// z=1: K = keys   @Wk -> Kh[h][S][64] bf16
// z=2: V = values @Wv -> Vt[h][64][S] bf16 (transposed for PV B-operand)
__global__ __launch_bounds__(128) void proj_qkv_kernel(
    const float* __restrict__ Xq, const float* __restrict__ Xk,
    const float* __restrict__ Xv, const __bf16* __restrict__ Wt,
    __bf16* __restrict__ Qh, __bf16* __restrict__ Kh, __bf16* __restrict__ Vt) {
  const int z = blockIdx.z;
  const float* X = (z == 0) ? Xq : (z == 1) ? Xk : Xv;
  const __bf16* W = Wt + (size_t)z * EMB * EMB;
  const int colb = blockIdx.x * 64;             // one full head per block col
  const int rowb = blockIdx.y * 64;
  const int wave = threadIdx.x >> 5;
  const int lane = threadIdx.x & 31;
  const int g = lane >> 4, n = lane & 15;
  const int row0 = rowb + wave * 16;

  v8f acc[4] = {};
  for (int kk = 0; kk < EMB; kk += 32) {
    v16bf a = a_frag_f32(X + (size_t)(row0 + n) * EMB + kk, g);
#pragma unroll
    for (int nt = 0; nt < 4; ++nt) {
      v16bf b = b_frag_bf16(W + (size_t)(colb + nt * 16 + n) * EMB + kk, g);
      acc[nt] = wmma_bf16(a, b, acc[nt]);
    }
  }
  const int h = colb >> 6;                      // head index (64-col aligned)
  if (z == 2) {
    // Vt[h][d][S]: element r runs over contiguous rows m -> one b128 per tile
#pragma unroll
    for (int nt = 0; nt < 4; ++nt) {
      v8bf pk;
#pragma unroll
      for (int r = 0; r < 8; ++r) pk[r] = (__bf16)acc[nt][r];
      *(v8bf*)(Vt + ((size_t)h * DH + nt * 16 + n) * S_LEN + row0 + 8 * g) = pk;
    }
  } else if (z == 1) {
#pragma unroll
    for (int nt = 0; nt < 4; ++nt)
#pragma unroll
      for (int r = 0; r < 8; ++r)
        Kh[((size_t)h * S_LEN + row0 + r + 8 * g) * DH + nt * 16 + n] =
            (__bf16)acc[nt][r];
  } else {
#pragma unroll
    for (int nt = 0; nt < 4; ++nt)
#pragma unroll
      for (int r = 0; r < 8; ++r)
        Qh[((size_t)h * S_LEN + row0 + r + 8 * g) * DH + nt * 16 + n] =
            (__bf16)acc[nt][r];
  }
}

// ---------------- Kernel 3: causal flash attention --------------------------
// one wave per (head, 16-query-row block); online softmax over key tiles of 32
__global__ __launch_bounds__(32) void attn_kernel(
    const __bf16* __restrict__ Qh, const __bf16* __restrict__ Kh,
    const __bf16* __restrict__ Vt, float* __restrict__ O) {
  __shared__ __align__(16) __bf16 Plds[16 * 32];   // P tile 16 rows x 32 keys
  const int h = blockIdx.y;
  const int q0 = blockIdx.x * 16;
  const int lane = threadIdx.x;
  const int g = lane >> 4, n = lane & 15;
  const __bf16* Q = Qh + (size_t)h * S_LEN * DH;
  const __bf16* K = Kh + (size_t)h * S_LEN * DH;
  const __bf16* V = Vt + (size_t)h * DH * S_LEN;

  const v16bf aq0 = a_frag_bf16(Q + (size_t)(q0 + n) * DH + 0, g);
  const v16bf aq1 = a_frag_bf16(Q + (size_t)(q0 + n) * DH + 32, g);

  float mrun[8], lrun[8];
  v8f acc[4] = {};
#pragma unroll
  for (int r = 0; r < 8; ++r) { mrun[r] = -1e30f; lrun[r] = 0.f; }

  const int jEnd = q0 + 16;                      // keys [0, row] allowed
  for (int j = 0; j < jEnd; j += 32) {
    // two 16x16 score tiles (keys j..j+15 and j+16..j+31), K depth 64
    v8f s0 = {}, s1 = {};
    {
      v16bf b00 = b_frag_bf16(K + (size_t)(j + n) * DH + 0, g);
      v16bf b01 = b_frag_bf16(K + (size_t)(j + n) * DH + 32, g);
      s0 = wmma_bf16(aq0, b00, s0);
      s0 = wmma_bf16(aq1, b01, s0);
      v16bf b10 = b_frag_bf16(K + (size_t)(j + 16 + n) * DH + 0, g);
      v16bf b11 = b_frag_bf16(K + (size_t)(j + 16 + n) * DH + 32, g);
      s1 = wmma_bf16(aq0, b10, s1);
      s1 = wmma_bf16(aq1, b11, s1);
    }
    const float sc = 0.125f;                     // 1/sqrt(64)
#pragma unroll
    for (int r = 0; r < 8; ++r) {
      const int m = q0 + r + 8 * g;              // query row of element r
      float v0 = s0[r] * sc;
      float v1 = s1[r] * sc;
      if (j + n > m) v0 = -1e30f;                // causal mask, col = j+16t+n
      if (j + 16 + n > m) v1 = -1e30f;
      float mx = fmaxf(v0, v1);
#pragma unroll
      for (int off = 1; off < 16; off <<= 1)
        mx = fmaxf(mx, __shfl_xor(mx, off, 32)); // row max within 16-lane half
      float mnew = fmaxf(mrun[r], mx);
      float p0 = __expf(v0 - mnew);
      float p1 = __expf(v1 - mnew);
      float sum = p0 + p1;
#pragma unroll
      for (int off = 1; off < 16; off <<= 1)
        sum += __shfl_xor(sum, off, 32);         // row sum
      float resc = __expf(mrun[r] - mnew);
      lrun[r] = lrun[r] * resc + sum;
      mrun[r] = mnew;
#pragma unroll
      for (int nt = 0; nt < 4; ++nt) acc[nt][r] *= resc;
      Plds[(r + 8 * g) * 32 + n] = (__bf16)p0;   // stage P in D-layout
      Plds[(r + 8 * g) * 32 + 16 + n] = (__bf16)p1;
    }
    __syncthreads();
    v16bf pa = a_frag_bf16(&Plds[n * 32], g);    // re-read P as A-fragment
#pragma unroll
    for (int nt = 0; nt < 4; ++nt) {
      v16bf bv = b_frag_bf16(V + (size_t)(nt * 16 + n) * S_LEN + j, g);
      acc[nt] = wmma_bf16(pa, bv, acc[nt]);
    }
    __syncthreads();
  }
  // epilogue: normalize, merge heads -> O[S][EMB] fp32
#pragma unroll
  for (int r = 0; r < 8; ++r) {
    float inv = 1.0f / lrun[r];
    int m = q0 + r + 8 * g;
#pragma unroll
    for (int nt = 0; nt < 4; ++nt)
      O[(size_t)m * EMB + h * DH + nt * 16 + n] = acc[nt][r] * inv;
  }
}

// ---------------- Kernel 4: output projection + bias ------------------------
__global__ __launch_bounds__(128) void proj_out_kernel(
    const float* __restrict__ Oin, const __bf16* __restrict__ WtO,
    const float* __restrict__ bias, float* __restrict__ Y) {
  const int colb = blockIdx.x * 64;
  const int rowb = blockIdx.y * 64;
  const int wave = threadIdx.x >> 5;
  const int lane = threadIdx.x & 31;
  const int g = lane >> 4, n = lane & 15;
  const int row0 = rowb + wave * 16;

  v8f acc[4] = {};
  for (int kk = 0; kk < EMB; kk += 32) {
    v16bf a = a_frag_f32(Oin + (size_t)(row0 + n) * EMB + kk, g);
#pragma unroll
    for (int nt = 0; nt < 4; ++nt) {
      v16bf b = b_frag_bf16(WtO + (size_t)(colb + nt * 16 + n) * EMB + kk, g);
      acc[nt] = wmma_bf16(a, b, acc[nt]);
    }
  }
#pragma unroll
  for (int nt = 0; nt < 4; ++nt) {
    int col = colb + nt * 16 + n;
    float bo = bias[col];
#pragma unroll
    for (int r = 0; r < 8; ++r) {
      int m = row0 + r + 8 * g;
      Y[(size_t)m * EMB + col] = acc[nt][r] + bo;
    }
  }
}

extern "C" void kernel_launch(void* const* d_in, const int* in_sizes, int n_in,
                              void* d_out, int out_size, void* d_ws, size_t ws_size,
                              hipStream_t stream) {
  const float* values  = (const float*)d_in[0];
  const float* keys    = (const float*)d_in[1];
  const float* queries = (const float*)d_in[2];
  // d_in[3] is the causal mask; we apply causality analytically instead.
  const float* Wq = (const float*)d_in[4];
  const float* Wk = (const float*)d_in[5];
  const float* Wv = (const float*)d_in[6];
  const float* Wo = (const float*)d_in[7];
  const float* bo = (const float*)d_in[8];
  float* Y = (float*)d_out;

  // workspace layout
  __bf16* Wt = (__bf16*)d_ws;                        // [4][EMB][EMB] bf16 (N-major)
  __bf16* Qh = Wt + (size_t)4 * EMB * EMB;           // [H][S][DH] bf16
  __bf16* Kh = Qh + (size_t)NH * S_LEN * DH;         // [H][S][DH] bf16
  __bf16* Vt = Kh + (size_t)NH * S_LEN * DH;         // [H][DH][S] bf16
  float*  Ob = (float*)(Vt + (size_t)NH * S_LEN * DH); // [S][EMB] f32

  transpose_w_kernel<<<dim3(EMB / 32, EMB / 32, 4), 256, 0, stream>>>(
      Wq, Wk, Wv, Wo, Wt);
  proj_qkv_kernel<<<dim3(EMB / 64, S_LEN / 64, 3), 128, 0, stream>>>(
      queries, keys, values, Wt, Qh, Kh, Vt);
  attn_kernel<<<dim3(S_LEN / 16, NH), 32, 0, stream>>>(Qh, Kh, Vt, Ob);
  proj_out_kernel<<<dim3(EMB / 64, S_LEN / 64), 128, 0, stream>>>(
      Ob, Wt + (size_t)3 * EMB * EMB, bo, Y);
}
